// simple_GCN_8701603741741
// MI455X (gfx1250) — compile-verified, matrix operand order checked by reading, not run on hardware
//
#include <hip/hip_runtime.h>
#include <hip/hip_bf16.h>
#include <stdint.h>

// ---------------------------------------------------------------------------
// 3-layer GCN + linear classifier for gfx1250 (MI455X).
//   h = relu( scatter_add( (h@W)[src] * norm, dst ) + b )   x3
//   out = h @ Wc + bc
// Dense parts: v_wmma_f32_16x16x32_bf16 (bf16 in, f32 acc), 5 M-tiles/block
// with B fragments register-resident across the M loop (amortizes W staging).
// Aggregation: one wave32 per edge, float4 gather + global f32 atomics.
// ---------------------------------------------------------------------------

#define N_NODES   50000
#define N_EDGES   800000
#define FEAT      128
#define N_CLS     40
#define N_CLS_PAD 48
#define MTILES    5                 // 16*5 = 80 rows per block; 50000/80 = 625

typedef __attribute__((ext_vector_type(16))) __bf16 v16bf;
typedef __attribute__((ext_vector_type(8)))  float  v8f;

union FragBF {
    v16bf v;
    uint4 q[2];
};

__device__ __forceinline__ unsigned short f2bf(float f) {
    // round-to-nearest-even fp32 -> bf16
    unsigned int u = __float_as_uint(f);
    unsigned int r = (u + 0x7FFFu + ((u >> 16) & 1u)) >> 16;
    return (unsigned short)r;
}

__device__ __forceinline__ unsigned int f2bf_pk(float lo, float hi) {
    return (unsigned int)f2bf(lo) | ((unsigned int)f2bf(hi) << 16);
}

// ---------------------------------------------------------------------------
// Degree / normalization
// ---------------------------------------------------------------------------
__global__ void k_deg_init(float* deg) {
    int i = blockIdx.x * blockDim.x + threadIdx.x;
    if (i < N_NODES) deg[i] = 1.0f;   // self loop contributes 1 to every node
}

__global__ void k_deg_edges(const int* __restrict__ ei, float* deg) {
    int e = blockIdx.x * blockDim.x + threadIdx.x;
    if (e < N_EDGES) atomicAdd(&deg[ei[N_EDGES + e]], 1.0f);  // dst-indexed
}

__global__ void k_dinv(float* deg) {
    int i = blockIdx.x * blockDim.x + threadIdx.x;
    if (i < N_NODES) {
        float d = deg[i];
        deg[i] = (d > 0.0f) ? __frsqrt_rn(d) : 0.0f;  // in place: deg -> dinv
    }
}

// ---------------------------------------------------------------------------
// Weight conversion fp32 -> bf16 (Wc padded 40 -> 48 cols with zeros)
// ---------------------------------------------------------------------------
__global__ void k_conv_weights(const float* __restrict__ W1,
                               const float* __restrict__ W2,
                               const float* __restrict__ W3,
                               const float* __restrict__ Wc,
                               unsigned short* __restrict__ Wb,   // 3 * 128*128
                               unsigned short* __restrict__ Wcb)  // 128*48
{
    int i = blockIdx.x * blockDim.x + threadIdx.x;
    const int WSZ = FEAT * FEAT;
    if (i < WSZ)            Wb[i]            = f2bf(W1[i]);
    else if (i < 2 * WSZ)   Wb[i]            = f2bf(W2[i - WSZ]);
    else if (i < 3 * WSZ)   Wb[i]            = f2bf(W3[i - 2 * WSZ]);
    else {
        int j = i - 3 * WSZ;
        if (j < FEAT * N_CLS_PAD) {
            int r = j / N_CLS_PAD, c = j % N_CLS_PAD;
            Wcb[j] = (c < N_CLS) ? f2bf(Wc[r * N_CLS + c]) : (unsigned short)0;
        }
    }
}

// ---------------------------------------------------------------------------
// WMMA GEMM: C[N_NODES x storeCols] = A[N_NODES x 128] @ Wb[128 x NCOL] (+bias)
// 256 threads = 8 waves; block owns 16*MTILES rows; wave w owns cols
// [16w,16w+16). W is staged transposed into LDS once per block; each wave
// loads its 4 B fragments (K=128) into registers ONCE and reuses them for all
// MTILES row-tiles. A tiles are fp32->bf16 converted into LDS; all fragment
// reads are contiguous 16B ds_load_b128 matching the 16-bit WMMA VGPR layout.
// ---------------------------------------------------------------------------
template <int NCOL>
__global__ __launch_bounds__(256)
void k_gemm_wmma(const float* __restrict__ A,
                 const unsigned short* __restrict__ Wb,  // [128][NCOL] bf16 row-major
                 const float* __restrict__ bias,         // [storeCols] or nullptr
                 float* __restrict__ C,
                 int ldC, int storeCols)
{
    __shared__ unsigned short Alds[16 * MTILES * FEAT];   // [row][k]
    __shared__ unsigned short Blds[NCOL * FEAT];          // transposed: [n][k]

    const int tid = threadIdx.x;
    const int m0  = blockIdx.x * (16 * MTILES);

    // Stage A tile (fp32 -> bf16, packed pair stores).  50000 % 80 == 0.
    {
        const float2* A2 = (const float2*)(A + (size_t)m0 * FEAT);
        unsigned int* Au = (unsigned int*)Alds;
        for (int i = tid; i < 16 * MTILES * FEAT / 2; i += 256) {
            float2 p = A2[i];
            Au[i] = f2bf_pk(p.x, p.y);
        }
    }
    // Stage W transposed: Blds[n*128 + k] = Wb[k*NCOL + n]
    for (int i = tid; i < FEAT * NCOL; i += 256) {
        int k = i / NCOL, n = i % NCOL;
        Blds[n * FEAT + k] = Wb[i];
    }
    __syncthreads();

    const int w    = tid >> 5;
    const int lane = tid & 31;
    if (w * 16 < NCOL) {
        const int rn  = lane & 15;           // A row within tile / B col within tile
        const int kb  = (lane >> 4) * 8;     // per-half-wave K base
        const int mro = (lane >> 4) * 8;     // C row offset within tile
        const int col = w * 16 + (lane & 15);

        // Hoist all 4 B fragments (full K=128) into registers.
        FragBF fb[4];
#pragma unroll
        for (int ks = 0; ks < 4; ++ks) {
            const unsigned short* bp = Blds + (w * 16 + rn) * FEAT + ks * 32 + kb;
            fb[ks].q[0] = *(const uint4*)(bp);
            fb[ks].q[1] = *(const uint4*)(bp + 16);
        }

        const float bv = (bias && col < storeCols) ? bias[col] : 0.0f;

#pragma unroll
        for (int mt = 0; mt < MTILES; ++mt) {
            v8f acc = {0.f, 0.f, 0.f, 0.f, 0.f, 0.f, 0.f, 0.f};
#pragma unroll
            for (int ks = 0; ks < 4; ++ks) {     // K = 128 in four 32-steps
                FragBF fa;
                const unsigned short* ap =
                    Alds + (mt * 16 + rn) * FEAT + ks * 32 + kb;
                fa.q[0] = *(const uint4*)(ap);        // K = ks*32+kb   .. +7
                fa.q[1] = *(const uint4*)(ap + 16);   // K = ks*32+16+kb.. +7
                acc = __builtin_amdgcn_wmma_f32_16x16x32_bf16(
                    false, fa.v, false, fb[ks].v, (short)0, acc, false, false);
            }
            // C/D layout: VGPR v -> row M = (lane>>4)*8 + v, col N = lane&15.
            if (col < storeCols) {
                const size_t rbase = (size_t)(m0 + mt * 16 + mro);
#pragma unroll
                for (int v = 0; v < 8; ++v) {
                    C[(rbase + v) * ldC + col] = acc[v] + bv;
                }
            }
        }
    }
}

// ---------------------------------------------------------------------------
// agg[i][f] = bias[f]   (pre-scatter init; reference adds bias after segsum)
// ---------------------------------------------------------------------------
__global__ void k_bias_init(float* __restrict__ agg, const float* __restrict__ b) {
    int i = blockIdx.x * blockDim.x + threadIdx.x;
    if (i < N_NODES * FEAT) agg[i] = b[i & 127];
}

// ---------------------------------------------------------------------------
// One wave32 per edge (plus N self-loop "edges"):
//   agg[dst] += t[src] * dinv[src]*dinv[dst]
// 32 lanes x float4 covers the 128-float feature row.
// ---------------------------------------------------------------------------
__global__ __launch_bounds__(256)
void k_aggregate(const float* __restrict__ t, const int* __restrict__ ei,
                 const float* __restrict__ dinv, float* __restrict__ agg)
{
    int gwid = (blockIdx.x * blockDim.x + threadIdx.x) >> 5;
    int lane = threadIdx.x & 31;
    if (gwid >= N_EDGES + N_NODES) return;

    int s, d;
    float nrm;
    if (gwid < N_EDGES) {
        s = ei[gwid];
        d = ei[N_EDGES + gwid];
        nrm = dinv[s] * dinv[d];
    } else {                       // self loop
        s = d = gwid - N_EDGES;
        float di = dinv[s];
        nrm = di * di;
    }

    float4 v = ((const float4*)(t + (size_t)s * FEAT))[lane];
    float* dp = agg + (size_t)d * FEAT + lane * 4;
    atomicAdd(dp + 0, v.x * nrm);
    atomicAdd(dp + 1, v.y * nrm);
    atomicAdd(dp + 2, v.z * nrm);
    atomicAdd(dp + 3, v.w * nrm);
}

__global__ void k_relu(float* __restrict__ h, int n) {
    int i = blockIdx.x * blockDim.x + threadIdx.x;
    if (i < n) h[i] = fmaxf(h[i], 0.0f);
}

// ---------------------------------------------------------------------------
// Host-side orchestration
// ---------------------------------------------------------------------------
extern "C" void kernel_launch(void* const* d_in, const int* in_sizes, int n_in,
                              void* d_out, int out_size, void* d_ws, size_t ws_size,
                              hipStream_t stream)
{
    (void)in_sizes; (void)n_in; (void)out_size; (void)ws_size;

    const float* x  = (const float*)d_in[0];
    const int*   ei = (const int*)  d_in[1];
    const float* W1 = (const float*)d_in[2];
    const float* b1 = (const float*)d_in[3];
    const float* W2 = (const float*)d_in[4];
    const float* b2 = (const float*)d_in[5];
    const float* W3 = (const float*)d_in[6];
    const float* b3 = (const float*)d_in[7];
    const float* Wc = (const float*)d_in[8];
    const float* bc = (const float*)d_in[9];

    // Workspace layout (byte offsets, 256B-aligned):
    char* ws = (char*)d_ws;
    float*          dinv = (float*)(ws + 0);                 // 50000 f  (deg -> dinv)
    unsigned short* Wb   = (unsigned short*)(ws + 200704);   // 3*16384 bf16
    unsigned short* Wcb  = (unsigned short*)(ws + 299008);   // 128*48  bf16
    float*          bufA = (float*)(ws + 311296);            // 6.4M f  (GEMM out)
    float*          bufB = (float*)(ws + 25911296);          // 6.4M f  (agg / h)

    float* out  = (float*)d_out;                  // [50000 x 40]
    float* hbuf = out + (size_t)N_NODES * N_CLS;  // [50000 x 128] final hidden

    const int HN  = N_NODES * FEAT;
    const int WSZ = FEAT * FEAT;

    // --- normalization ---
    k_deg_init <<<(N_NODES + 255) / 256, 256, 0, stream>>>(dinv);
    k_deg_edges<<<(N_EDGES + 255) / 256, 256, 0, stream>>>(ei, dinv);
    k_dinv     <<<(N_NODES + 255) / 256, 256, 0, stream>>>(dinv);

    // --- weights to bf16 ---
    {
        int tot = 3 * WSZ + FEAT * N_CLS_PAD;
        k_conv_weights<<<(tot + 255) / 256, 256, 0, stream>>>(W1, W2, W3, Wc, Wb, Wcb);
    }

    const int gemmGrid = N_NODES / (16 * MTILES);          // 625, exact
    const int aggGrid  = (N_EDGES + N_NODES + 7) / 8;      // 8 waves / block
    const int elGrid   = (HN + 255) / 256;

    // --- layer 1 ---
    k_gemm_wmma<FEAT><<<gemmGrid, 256, 0, stream>>>(x, Wb, nullptr, bufA, FEAT, FEAT);
    k_bias_init<<<elGrid, 256, 0, stream>>>(bufB, b1);
    k_aggregate<<<aggGrid, 256, 0, stream>>>(bufA, ei, dinv, bufB);
    k_relu<<<elGrid, 256, 0, stream>>>(bufB, HN);

    // --- layer 2 ---
    k_gemm_wmma<FEAT><<<gemmGrid, 256, 0, stream>>>(bufB, Wb + WSZ, nullptr, bufA, FEAT, FEAT);
    k_bias_init<<<elGrid, 256, 0, stream>>>(bufB, b2);
    k_aggregate<<<aggGrid, 256, 0, stream>>>(bufA, ei, dinv, bufB);
    k_relu<<<elGrid, 256, 0, stream>>>(bufB, HN);

    // --- layer 3 (hidden lands directly in d_out's h region) ---
    k_gemm_wmma<FEAT><<<gemmGrid, 256, 0, stream>>>(bufB, Wb + 2 * WSZ, nullptr, bufA, FEAT, FEAT);
    k_bias_init<<<elGrid, 256, 0, stream>>>(hbuf, b3);
    k_aggregate<<<aggGrid, 256, 0, stream>>>(bufA, ei, dinv, hbuf);
    k_relu<<<elGrid, 256, 0, stream>>>(hbuf, HN);

    // --- classifier: out = h @ Wc + bc (48-padded tiles, store 40 cols) ---
    k_gemm_wmma<N_CLS_PAD><<<gemmGrid, 256, 0, stream>>>(hbuf, Wcb, bc, out, N_CLS, N_CLS);
}
